// myRNN_18674517803367
// MI455X (gfx1250) — matrix operand order repaired and checked by reading, not compiled, not run on hardware
//
#include <hip/hip_runtime.h>
#include <math.h>

#define B_  256
#define S_  512
#define E_  300
#define EP_ 320      // E padded to multiple of 32
#define H_  1024
#define KC_ 1344     // EP_ + H_
#define H2_ 2048

typedef __attribute__((ext_vector_type(16))) __bf16 v16bf;
typedef __attribute__((ext_vector_type(8)))  __bf16 v8bf;
typedef __attribute__((ext_vector_type(8)))  float  v8f;

struct BfPair { v8bf lo, hi; };

// Load a 16-element bf16 fragment as two 16-byte chunks (p and p+off2 elems).
__device__ __forceinline__ v16bf ldfrag(const __bf16* p, int off2) {
  BfPair t;
  t.lo = *(const v8bf*)(p);
  t.hi = *(const v8bf*)(p + off2);
  return __builtin_bit_cast(v16bf, t);
}

// Accumulate a [rowBase..+32) x [colBase..+32) tile over kLen of K.
// A: row-major [M][sA] activations. Bw: row-major weights, row n holds W[n][k]
// (so it acts as B^T for D = A * B). K order used here (same for A and B, not the ISA table order):
//  A frag (16x32): lane l16 = row, elems 0..15 = K[half*16..+15] contiguous.
//  B frag (32x16): lane l16 = col, elems 0..15 = K[half*16..+15] contiguous.
__device__ __forceinline__ void mma_phase(v8f (&acc)[2][2],
    const __bf16* __restrict__ A, int sA,
    const __bf16* __restrict__ Bw, int sB,
    int kLen, int rowBase, int colBase) {
  const int lane = threadIdx.x & 31;
  const int l16 = lane & 15, half = lane >> 4;
  const __bf16* a0 = A + (size_t)(rowBase + l16) * sA + half * 16;
  const __bf16* a1 = a0 + (size_t)16 * sA;
  const __bf16* b0 = Bw + (size_t)(colBase + l16) * sB + half * 16;
  const __bf16* b1 = b0 + (size_t)16 * sB;
  for (int k = 0; k < kLen; k += 32) {
    v16bf fa0 = ldfrag(a0 + k, 8);
    v16bf fa1 = ldfrag(a1 + k, 8);
    v16bf fb0 = ldfrag(b0 + k, 8);
    v16bf fb1 = ldfrag(b1 + k, 8);
    acc[0][0] = __builtin_amdgcn_wmma_f32_16x16x32_bf16(false, fa0, false, fb0, (short)0, acc[0][0], false, false);
    acc[0][1] = __builtin_amdgcn_wmma_f32_16x16x32_bf16(false, fa0, false, fb1, (short)0, acc[0][1], false, false);
    acc[1][0] = __builtin_amdgcn_wmma_f32_16x16x32_bf16(false, fa1, false, fb0, (short)0, acc[1][0], false, false);
    acc[1][1] = __builtin_amdgcn_wmma_f32_16x16x32_bf16(false, fa1, false, fb1, (short)0, acc[1][1], false, false);
  }
}

// ---------------- persistent recurrence ----------------
// grid (4,8) = 32 blocks x 256 threads (8 wave32). Block tile 64x128,
// wave tile 32x32 (2x2 WMMA tiles). Device-wide barrier between steps
// via monotonically increasing atomic counter.
__global__ __launch_bounds__(256) void rnn_persistent(
    const unsigned short* __restrict__ Ebf_,
    unsigned short* __restrict__ H0_, unsigned short* __restrict__ H1_,
    const unsigned short* __restrict__ Wcat_,
    const float* __restrict__ bih,
    unsigned* __restrict__ counter) {
  const __bf16* E = (const __bf16*)Ebf_;
  const __bf16* W = (const __bf16*)Wcat_;
  __bf16* Hb0 = (__bf16*)H0_;
  __bf16* Hb1 = (__bf16*)H1_;

  const int wave  = threadIdx.x >> 5;
  const int waveM = wave >> 2, waveN = wave & 3;
  const int rowBase = blockIdx.x * 64  + waveM * 32;
  const int colBase = blockIdx.y * 128 + waveN * 32;
  const int lane = threadIdx.x & 31;
  const int l16 = lane & 15, half = lane >> 4;
  const unsigned nBlocks = gridDim.x * gridDim.y;

  for (int t = 0; t < S_; ++t) {
    const __bf16* Hin  = (t & 1) ? Hb1 : Hb0;
    __bf16*       Hout = (t & 1) ? Hb0 : Hb1;
    v8f acc[2][2] = {};
    // phase 1: K over embedding columns (Wcat cols [0,320))
    mma_phase(acc, E + (size_t)t * B_ * EP_, EP_, W, KC_, EP_, rowBase, colBase);
    // phase 2: K over hidden state (Wcat cols [320,1344))
    mma_phase(acc, Hin, H_, W + EP_, KC_, H_, rowBase, colBase);

    // epilogue: D layout -> lane l16 = col n, vgpr r -> row m = r + half*8
    #pragma unroll
    for (int tm = 0; tm < 2; ++tm)
      #pragma unroll
      for (int tn = 0; tn < 2; ++tn) {
        const int col = colBase + tn * 16 + l16;
        const float bias = bih[col];
        const int m0 = rowBase + tm * 16 + half * 8;
        #pragma unroll
        for (int r = 0; r < 8; ++r) {
          float v = tanhf(acc[tm][tn][r] + bias);
          Hout[(size_t)(m0 + r) * H_ + col] = (__bf16)v;
        }
      }

    // device-wide barrier: make stores visible, then count arrivals
    __threadfence();
    __syncthreads();
    if (threadIdx.x == 0) {
      atomicAdd(counter, 1u);
      const unsigned target = nBlocks * (unsigned)(t + 1);
      while (__hip_atomic_load(counter, __ATOMIC_ACQUIRE, __HIP_MEMORY_SCOPE_AGENT) < target)
        __builtin_amdgcn_s_sleep(1);
    }
    __syncthreads();
  }
}

// ---------------- head: hid = relu(h_last @ W1^T + b1) ----------------
__global__ __launch_bounds__(256) void head_gemm(
    const unsigned short* __restrict__ Hlast_,
    const unsigned short* __restrict__ W1bf_,
    const float* __restrict__ b1,
    float* __restrict__ hid) {
  const __bf16* A  = (const __bf16*)Hlast_;
  const __bf16* Bw = (const __bf16*)W1bf_;
  const int wave  = threadIdx.x >> 5;
  const int waveM = wave >> 2, waveN = wave & 3;
  const int rowBase = blockIdx.x * 64  + waveM * 32;
  const int colBase = blockIdx.y * 128 + waveN * 32;
  const int lane = threadIdx.x & 31;
  const int l16 = lane & 15, half = lane >> 4;

  v8f acc[2][2] = {};
  mma_phase(acc, A, H_, Bw, H_, H_, rowBase, colBase);

  #pragma unroll
  for (int tm = 0; tm < 2; ++tm)
    #pragma unroll
    for (int tn = 0; tn < 2; ++tn) {
      const int col = colBase + tn * 16 + l16;
      const float bias = b1[col];
      const int m0 = rowBase + tm * 16 + half * 8;
      #pragma unroll
      for (int r = 0; r < 8; ++r) {
        float v = acc[tm][tn][r] + bias;
        hid[(size_t)(m0 + r) * H2_ + col] = v > 0.0f ? v : 0.0f;
      }
    }
}

// ---------------- out[b] = hid[b] . W2 + b2 ----------------
__global__ __launch_bounds__(256) void out_dot(const float* __restrict__ hid,
    const float* __restrict__ W2, const float* __restrict__ b2,
    float* __restrict__ out) {
  const int b = blockIdx.x;
  const float* row = hid + (size_t)b * H2_;
  float s = 0.0f;
  for (int k = threadIdx.x; k < H2_; k += 256) s += row[k] * W2[k];
  __shared__ float red[256];
  red[threadIdx.x] = s;
  __syncthreads();
  for (int off = 128; off > 0; off >>= 1) {
    if ((int)threadIdx.x < off) red[threadIdx.x] += red[threadIdx.x + off];
    __syncthreads();
  }
  if (threadIdx.x == 0) out[b] = red[0] + b2[0];
}

// ---------------- packing / init ----------------
// Wcat[h][0..299]=W_e[h], [300..319]=0, [320..1343]=W_h[h]; row-major bf16.
__global__ void pack_wcat(const float* __restrict__ W_ih,
                          unsigned short* __restrict__ Wcat_) {
  __bf16* Wcat = (__bf16*)Wcat_;
  int i = blockIdx.x * 256 + threadIdx.x;
  if (i >= H_ * KC_) return;
  int h = i / KC_, c = i % KC_;
  float v = 0.0f;
  if (c < E_)        v = W_ih[(size_t)h * (E_ + H_) + c];
  else if (c >= EP_) v = W_ih[(size_t)h * (E_ + H_) + E_ + (c - EP_)];
  Wcat[i] = (__bf16)v;
}

__global__ void pack_w1(const float* __restrict__ W1,
                        unsigned short* __restrict__ W1bf_) {
  __bf16* W1bf = (__bf16*)W1bf_;
  int i = blockIdx.x * 256 + threadIdx.x;
  if (i < H2_ * H_) W1bf[i] = (__bf16)W1[i];
}

// Ebf[t][b][c] = bf16(embed_W[x[b][t]][c]) for c<300, else 0.
__global__ void gather_embed(const int* __restrict__ x,
                             const float* __restrict__ embW,
                             unsigned short* __restrict__ Ebf_) {
  __bf16* Ebf = (__bf16*)Ebf_;
  int i = blockIdx.x * 256 + threadIdx.x;
  if (i >= S_ * B_ * EP_) return;
  int c  = i % EP_;
  int bt = i / EP_;
  int b  = bt % B_;
  int t  = bt / B_;
  float v = 0.0f;
  if (c < E_) {
    int row = x[(size_t)b * S_ + t];
    v = embW[(size_t)row * E_ + c];
  }
  Ebf[i] = (__bf16)v;
}

__global__ void init_state(unsigned short* __restrict__ H0_,
                           unsigned* __restrict__ counter) {
  int i = blockIdx.x * 256 + threadIdx.x;
  if (i == 0) *counter = 0u;
  if (i < B_ * H_) ((__bf16*)H0_)[i] = (__bf16)0.0f;
}

extern "C" void kernel_launch(void* const* d_in, const int* in_sizes, int n_in,
                              void* d_out, int out_size, void* d_ws, size_t ws_size,
                              hipStream_t stream) {
  const int*   x    = (const int*)  d_in[0];
  const float* embW = (const float*)d_in[1];
  const float* W_ih = (const float*)d_in[2];
  const float* bih  = (const float*)d_in[3];
  const float* W1   = (const float*)d_in[4];
  const float* b1   = (const float*)d_in[5];
  const float* W2   = (const float*)d_in[6];
  const float* b2   = (const float*)d_in[7];
  float* out = (float*)d_out;

  char* p = (char*)d_ws;
  auto take = [&p](size_t bytes) {
    char* r = p;
    p += (bytes + 255) & ~(size_t)255;
    return r;
  };
  unsigned short* Wcat = (unsigned short*)take((size_t)H_ * KC_ * 2);   // 2.75 MB
  unsigned short* W1bf = (unsigned short*)take((size_t)H2_ * H_ * 2);   // 4 MB
  unsigned short* Ebf  = (unsigned short*)take((size_t)S_ * B_ * EP_ * 2); // 80 MB
  unsigned short* Hb0  = (unsigned short*)take((size_t)B_ * H_ * 2);
  unsigned short* Hb1  = (unsigned short*)take((size_t)B_ * H_ * 2);
  float*          hid  = (float*)take((size_t)B_ * H2_ * 4);            // 2 MB
  unsigned*       cnt  = (unsigned*)take(256);

  pack_wcat   <<<(H_ * KC_ + 255) / 256,      256, 0, stream>>>(W_ih, Wcat);
  pack_w1     <<<(H2_ * H_ + 255) / 256,      256, 0, stream>>>(W1, W1bf);
  gather_embed<<<(S_ * B_ * EP_ + 255) / 256, 256, 0, stream>>>(x, embW, Ebf);
  init_state  <<<(B_ * H_ + 255) / 256,       256, 0, stream>>>(Hb0, cnt);

  // 32 co-resident blocks run all 512 steps with an internal global barrier.
  rnn_persistent<<<dim3(B_ / 64, H_ / 128), 256, 0, stream>>>(Ebf, Hb0, Hb1, Wcat, bih, cnt);
  // final h is in Hb0 after an even number (512) of steps.
  head_gemm<<<dim3(B_ / 64, H2_ / 128), 256, 0, stream>>>(Hb0, W1bf, b1, hid);
  out_dot  <<<B_, 256, 0, stream>>>(hid, W2, b2, out);

  (void)in_sizes; (void)n_in; (void)out_size; (void)ws_size;
}
